// TransformerLayer_876173328474
// MI455X (gfx1250) — matrix operand order repaired and checked by reading, not compiled, not run on hardware
//
#include <hip/hip_runtime.h>
#include <hip/hip_bf16.h>
#include <math.h>

// Problem constants (match reference)
#define B_   16
#define N_   400
#define D_   256
#define H_   8
#define DH_  32
#define NW_  4
#define FF_  1024
#define QK_SCALE 0.17677669529663687f  // 1/sqrt(32)

typedef __attribute__((ext_vector_type(16))) __bf16 v16bf;
typedef __attribute__((ext_vector_type(8)))  __bf16 bf8;
typedef __attribute__((ext_vector_type(8)))  float  v8f;

// ---- fragment-load helpers -------------------------------------------------

static __device__ inline bf8 ld8_bf(const __bf16* p) {
  return *(const bf8*)p;  // 16B contiguous bf16 load (alignment guaranteed by layout)
}

static __device__ inline bf8 ld8_cvt(const float* p) {
  float4 a = *(const float4*)p;
  float4 b = *(const float4*)(p + 4);
  bf8 r;
  r[0] = (__bf16)a.x; r[1] = (__bf16)a.y; r[2] = (__bf16)a.z; r[3] = (__bf16)a.w;
  r[4] = (__bf16)b.x; r[5] = (__bf16)b.y; r[6] = (__bf16)b.z; r[7] = (__bf16)b.w;
  return r;
}

static __device__ inline bf8 zero8() {
  bf8 r;
#pragma unroll
  for (int i = 0; i < 8; ++i) r[i] = (__bf16)0.0f;
  return r;
}

static __device__ inline v16bf cat16(bf8 lo, bf8 hi) {
  return __builtin_shufflevector(lo, hi, 0,1,2,3,4,5,6,7,8,9,10,11,12,13,14,15);
}

static __device__ inline v8f wmma_bf16(v16bf a, v16bf b, v8f c) {
  // D(16x16,f32) = A(16x32,bf16) * B(32x16,bf16) + C
  return __builtin_amdgcn_wmma_f32_16x16x32_bf16(false, a, false, b, (short)0, c,
                                                 false, false);
}

// A fragment: row m = lane&15; elems 0..7 at K=k0+half*8, 8..15 at K=k0+16+half*8
// B fragment: col n = lane&15; elems 0..7 at K=k0+half*8, 8..15 at K=k0+16+half*8

// ---- kernel 0: one-time fp32 -> bf16 tensor conversion ---------------------
__global__ void __launch_bounds__(256)
cvt_kernel(const float* __restrict__ in, __bf16* __restrict__ out, int n8)
{
  int i = blockIdx.x * 256 + threadIdx.x;
  if (i < n8) {
    bf8 v = ld8_cvt(in + (size_t)i * 8);
    *(bf8*)(out + (size_t)i * 8) = v;
  }
}

// ---- kernel 1: fused QKV projections (z = 0/1/2 -> q/k/v) ------------------
// Each wave: 16x64 output tile (4 accumulators reuse the A fragment).
__global__ void __launch_bounds__(32)
qkv_kernel(const __bf16* __restrict__ xb,
           const __bf16* __restrict__ Wqb, const float* __restrict__ bq,
           const __bf16* __restrict__ Wkb, const float* __restrict__ bk,
           const __bf16* __restrict__ Wvb, const float* __restrict__ bv,
           __bf16* __restrict__ qb, __bf16* __restrict__ kb,
           __bf16* __restrict__ vT)
{
  const int lane = threadIdx.x, half = lane >> 4, lm = lane & 15;
  const int m0 = blockIdx.x * 16, ng = blockIdx.y, z = blockIdx.z;
  const __bf16* W   = (z == 0) ? Wqb : (z == 1) ? Wkb : Wvb;
  const float* bias = (z == 0) ? bq : (z == 1) ? bk : bv;
  const __bf16* rowA = xb + (size_t)(m0 + lm) * D_;
  const __bf16* rowB[4];
#pragma unroll
  for (int j = 0; j < 4; ++j)
    rowB[j] = W + (size_t)(ng * 64 + j * 16 + lm) * D_;
  v8f c[4] = {};
#pragma unroll
  for (int k0 = 0; k0 < D_; k0 += 32) {
    v16bf a = cat16(ld8_bf(rowA + k0 + half * 8),
                    ld8_bf(rowA + k0 + 16 + half * 8));
#pragma unroll
    for (int j = 0; j < 4; ++j) {
      v16bf b = cat16(ld8_bf(rowB[j] + k0 + half * 8),
                      ld8_bf(rowB[j] + k0 + 16 + half * 8));
      c[j] = wmma_bf16(a, b, c[j]);
    }
  }
#pragma unroll
  for (int j = 0; j < 4; ++j) {
    const int n = ng * 64 + j * 16 + lm, h = n >> 5, dh = n & 31;
    const float bia = bias[n];
#pragma unroll
    for (int r = 0; r < 8; ++r) {
      int mrow = m0 + half * 8 + r;
      int bb = mrow / N_, seq = mrow - bb * N_;
      float val = c[j][r] + bia;
      if (z == 0)
        qb[((size_t)(bb * H_ + h) * N_ + seq) * DH_ + dh] = (__bf16)(val * QK_SCALE);
      else if (z == 1)
        kb[((size_t)(bb * H_ + h) * N_ + seq) * DH_ + dh] = (__bf16)val;
      else
        vT[((size_t)(bb * H_ + h) * DH_ + dh) * N_ + seq] = (__bf16)val;  // V^T
    }
  }
}

// ---- kernel 2: qk = (q/sqrt(dh)) @ k^T, into d_out scores region -----------
// Each wave: 16x80 tile (5 accumulators, one K=32 step, A loaded once).
__global__ void __launch_bounds__(32)
qk_kernel(const __bf16* __restrict__ qb, const __bf16* __restrict__ kb,
          float* __restrict__ qk)
{
  const int lane = threadIdx.x, half = lane >> 4, lm = lane & 15;
  const int bh = blockIdx.y;
  const int nt = blockIdx.x / 5, mg = blockIdx.x % 5;  // 25 n-tiles x 5 m-groups
  const int n0 = nt * 16;
  const __bf16* rowA = qb + ((size_t)bh * N_ + n0 + lm) * DH_;
  v16bf a = cat16(ld8_bf(rowA + half * 8), ld8_bf(rowA + 16 + half * 8));
  v8f c[5];
#pragma unroll
  for (int j = 0; j < 5; ++j) {
    const __bf16* rowB = kb + ((size_t)bh * N_ + mg * 80 + j * 16 + lm) * DH_;
    v16bf b = cat16(ld8_bf(rowB + half * 8), ld8_bf(rowB + 16 + half * 8));
    v8f z = {};
    c[j] = wmma_bf16(a, b, z);
  }
  float* orow = qk + (size_t)bh * N_ * N_;
#pragma unroll
  for (int j = 0; j < 5; ++j)
#pragma unroll
    for (int r = 0; r < 8; ++r)
      orow[(size_t)(n0 + half * 8 + r) * N_ + mg * 80 + j * 16 + lm] = c[j][r];
}

// ---- kernel 3: signed-softmax scores, in place over d_out ------------------
__global__ void __launch_bounds__(256)
scores_kernel(float* __restrict__ qk, const float* __restrict__ aw)
{
  const int n = blockIdx.x, b = blockIdx.y, tid = threadIdx.x;
  const int c0 = tid, c1 = tid + 256;
  const bool has1 = (c1 < N_);
  float w0[NW_], w1[NW_];
#pragma unroll
  for (int i = 0; i < NW_; ++i) {
    const float* wrow = aw + (((size_t)b * NW_ + i) * N_ + n) * N_;
    w0[i] = wrow[c0];
    w1[i] = has1 ? wrow[c1] : 0.0f;
  }
  __shared__ float red[256];
  for (int h = 0; h < H_; ++h) {
    float* row = qk + (((size_t)b * H_ + h) * N_ + n) * N_;
    float q0 = row[c0];
    float q1 = has1 ? row[c1] : 0.0f;
    float a0 = 0.f, a1 = 0.f;
#pragma unroll
    for (int i = 0; i < NW_; ++i) {
      float s0 = q0 * w0[i], s1 = q1 * w1[i];
      float e0 = __expf(fabsf(s0));
      float e1 = has1 ? __expf(fabsf(s1)) : 0.0f;
      red[tid] = e0 + e1;
      __syncthreads();
      for (int s = 128; s > 0; s >>= 1) {
        if (tid < s) red[tid] += red[tid + s];
        __syncthreads();
      }
      float inv = 1.0f / (red[0] + 1e-10f);
      __syncthreads();
      float g0 = (s0 > 0.f) ? 1.f : (s0 < 0.f ? -1.f : 0.f);
      float g1 = (s1 > 0.f) ? 1.f : (s1 < 0.f ? -1.f : 0.f);
      a0 += g0 * e0 * inv;
      a1 += g1 * e1 * inv;
    }
    row[c0] = a0 * 0.25f;
    if (has1) row[c1] = a1 * 0.25f;
  }
}

// ---- kernel 4: attn_out = scores @ v ---------------------------------------
// Each wave: 16x32 tile (both dh-tiles), K=400: 12 full steps + K=16 tail.
__global__ void __launch_bounds__(32)
av_kernel(const float* __restrict__ sc, const __bf16* __restrict__ vT,
          __bf16* __restrict__ att)
{
  const int lane = threadIdx.x, half = lane >> 4, lm = lane & 15;
  const int bh = blockIdx.y, b = bh >> 3, h = bh & 7;
  const int n0 = blockIdx.x * 16;
  const float* rowA = sc + ((size_t)bh * N_ + n0 + lm) * N_;
  const __bf16* rowB0 = vT + ((size_t)bh * DH_ + lm) * N_;
  const __bf16* rowB1 = vT + ((size_t)bh * DH_ + 16 + lm) * N_;
  v8f c0 = {}, c1 = {};
  for (int k0 = 0; k0 < 384; k0 += 32) {
    __builtin_prefetch(rowA + k0 + 64, 0, 1);
    v16bf a = cat16(ld8_cvt(rowA + k0 + half * 8),
                    ld8_cvt(rowA + k0 + 16 + half * 8));
    v16bf b0 = cat16(ld8_bf(rowB0 + k0 + half * 8),
                     ld8_bf(rowB0 + k0 + 16 + half * 8));
    v16bf b1 = cat16(ld8_bf(rowB1 + k0 + half * 8),
                     ld8_bf(rowB1 + k0 + 16 + half * 8));
    c0 = wmma_bf16(a, b0, c0);
    c1 = wmma_bf16(a, b1, c1);
  }
  {  // tail K=384..399, zero-padded to 32
    v16bf a = cat16(ld8_cvt(rowA + 384 + half * 8), zero8());
    v16bf b0 = cat16(ld8_bf(rowB0 + 384 + half * 8), zero8());
    v16bf b1 = cat16(ld8_bf(rowB1 + 384 + half * 8), zero8());
    c0 = wmma_bf16(a, b0, c0);
    c1 = wmma_bf16(a, b1, c1);
  }
#pragma unroll
  for (int r = 0; r < 8; ++r) {
    int seq = n0 + half * 8 + r;
    __bf16* orow = att + ((size_t)b * N_ + seq) * D_ + h * DH_;
    orow[lm]      = (__bf16)c0[r];
    orow[16 + lm] = (__bf16)c1[r];
  }
}

// ---- kernel 5: tmp1 = att @ Wo^T + bo + x (16x64 tile per wave) ------------
__global__ void __launch_bounds__(32)
oproj_kernel(const __bf16* __restrict__ att, const __bf16* __restrict__ Wob,
             const float* __restrict__ bo, const float* __restrict__ x,
             float* __restrict__ tmp)
{
  const int lane = threadIdx.x, half = lane >> 4, lm = lane & 15;
  const int m0 = blockIdx.x * 16, ng = blockIdx.y;
  const __bf16* rowA = att + (size_t)(m0 + lm) * D_;
  const __bf16* rowB[4];
#pragma unroll
  for (int j = 0; j < 4; ++j)
    rowB[j] = Wob + (size_t)(ng * 64 + j * 16 + lm) * D_;
  v8f c[4] = {};
#pragma unroll
  for (int k0 = 0; k0 < D_; k0 += 32) {
    v16bf a = cat16(ld8_bf(rowA + k0 + half * 8),
                    ld8_bf(rowA + k0 + 16 + half * 8));
#pragma unroll
    for (int j = 0; j < 4; ++j) {
      v16bf b = cat16(ld8_bf(rowB[j] + k0 + half * 8),
                      ld8_bf(rowB[j] + k0 + 16 + half * 8));
      c[j] = wmma_bf16(a, b, c[j]);
    }
  }
#pragma unroll
  for (int j = 0; j < 4; ++j) {
    const int n = ng * 64 + j * 16 + lm;
    const float bia = bo[n];
#pragma unroll
    for (int r = 0; r < 8; ++r) {
      size_t idx = (size_t)(m0 + half * 8 + r) * D_ + n;
      tmp[idx] = c[j][r] + bia + x[idx];
    }
  }
}

// ---- layer norm (writes f32 and optionally bf16 copy) ----------------------
__global__ void __launch_bounds__(256)
ln_kernel(const float* __restrict__ in, const float* __restrict__ g,
          const float* __restrict__ bt, float* __restrict__ outf,
          __bf16* __restrict__ outb)
{
  const int tid = threadIdx.x;
  const size_t base = (size_t)blockIdx.x * D_;
  float t = in[base + tid];
  __shared__ float red[256];
  red[tid] = t;
  __syncthreads();
  for (int s = 128; s > 0; s >>= 1) {
    if (tid < s) red[tid] += red[tid + s];
    __syncthreads();
  }
  float mean = red[0] * (1.0f / D_);
  __syncthreads();
  float d = t - mean;
  red[tid] = d * d;
  __syncthreads();
  for (int s = 128; s > 0; s >>= 1) {
    if (tid < s) red[tid] += red[tid + s];
    __syncthreads();
  }
  float var = red[0] * (1.0f / D_);
  float val = d * rsqrtf(var + 1e-5f) * g[tid] + bt[tid];
  if (outf) outf[base + tid] = val;
  if (outb) outb[base + tid] = (__bf16)val;
}

// ---- kernel 7: r = relu(h @ W1^T + b1) (16x64 tile per wave) ---------------
__global__ void __launch_bounds__(32)
ffn1_kernel(const __bf16* __restrict__ hb, const __bf16* __restrict__ W1b,
            const float* __restrict__ b1, __bf16* __restrict__ rb)
{
  const int lane = threadIdx.x, half = lane >> 4, lm = lane & 15;
  const int m0 = blockIdx.x * 16, ng = blockIdx.y;
  const __bf16* rowA = hb + (size_t)(m0 + lm) * D_;
  const __bf16* rowB[4];
#pragma unroll
  for (int j = 0; j < 4; ++j)
    rowB[j] = W1b + (size_t)(ng * 64 + j * 16 + lm) * D_;
  v8f c[4] = {};
#pragma unroll
  for (int k0 = 0; k0 < D_; k0 += 32) {
    v16bf a = cat16(ld8_bf(rowA + k0 + half * 8),
                    ld8_bf(rowA + k0 + 16 + half * 8));
#pragma unroll
    for (int j = 0; j < 4; ++j) {
      v16bf b = cat16(ld8_bf(rowB[j] + k0 + half * 8),
                      ld8_bf(rowB[j] + k0 + 16 + half * 8));
      c[j] = wmma_bf16(a, b, c[j]);
    }
  }
#pragma unroll
  for (int j = 0; j < 4; ++j) {
    const int n = ng * 64 + j * 16 + lm;
    const float bia = b1[n];
#pragma unroll
    for (int r = 0; r < 8; ++r) {
      float v = c[j][r] + bia;
      rb[(size_t)(m0 + half * 8 + r) * FF_ + n] = (__bf16)(v > 0.f ? v : 0.f);
    }
  }
}

// ---- kernel 8: tmp2 = r @ W2^T + b2 + h (16x64 tile per wave) --------------
__global__ void __launch_bounds__(32)
ffn2_kernel(const __bf16* __restrict__ rb, const __bf16* __restrict__ W2b,
            const float* __restrict__ b2, const float* __restrict__ hf,
            float* __restrict__ tmp2)
{
  const int lane = threadIdx.x, half = lane >> 4, lm = lane & 15;
  const int m0 = blockIdx.x * 16, ng = blockIdx.y;
  const __bf16* rowA = rb + (size_t)(m0 + lm) * FF_;
  const __bf16* rowB[4];
#pragma unroll
  for (int j = 0; j < 4; ++j)
    rowB[j] = W2b + (size_t)(ng * 64 + j * 16 + lm) * FF_;
  v8f c[4] = {};
  for (int k0 = 0; k0 < FF_; k0 += 32) {
    __builtin_prefetch(rowA + k0 + 64, 0, 1);
    v16bf a = cat16(ld8_bf(rowA + k0 + half * 8),
                    ld8_bf(rowA + k0 + 16 + half * 8));
#pragma unroll
    for (int j = 0; j < 4; ++j) {
      v16bf b = cat16(ld8_bf(rowB[j] + k0 + half * 8),
                      ld8_bf(rowB[j] + k0 + 16 + half * 8));
      c[j] = wmma_bf16(a, b, c[j]);
    }
  }
#pragma unroll
  for (int j = 0; j < 4; ++j) {
    const int n = ng * 64 + j * 16 + lm;
    const float bia = b2[n];
#pragma unroll
    for (int r = 0; r < 8; ++r) {
      size_t idx = (size_t)(m0 + half * 8 + r) * D_ + n;
      tmp2[idx] = c[j][r] + bia + hf[idx];
    }
  }
}

// ---- launcher --------------------------------------------------------------
extern "C" void kernel_launch(void* const* d_in, const int* in_sizes, int n_in,
                              void* d_out, int out_size, void* d_ws, size_t ws_size,
                              hipStream_t stream)
{
  const float* x   = (const float*)d_in[0];
  const float* aw  = (const float*)d_in[1];
  const float* Wq  = (const float*)d_in[2];
  const float* bq  = (const float*)d_in[3];
  const float* Wk  = (const float*)d_in[4];
  const float* bk  = (const float*)d_in[5];
  const float* Wv  = (const float*)d_in[6];
  const float* bv  = (const float*)d_in[7];
  const float* Wo  = (const float*)d_in[8];
  const float* bo  = (const float*)d_in[9];
  const float* g1  = (const float*)d_in[10];
  const float* be1 = (const float*)d_in[11];
  const float* W1  = (const float*)d_in[12];
  const float* b1  = (const float*)d_in[13];
  const float* W2  = (const float*)d_in[14];
  const float* b2  = (const float*)d_in[15];
  const float* g2  = (const float*)d_in[16];
  const float* be2 = (const float*)d_in[17];

  float* y_out  = (float*)d_out;
  float* sc_out = y_out + (size_t)B_ * N_ * D_;  // second output: scores [B,H,N,N]

  // workspace carve-up (~54 MB)
  char* p = (char*)d_ws;
  auto take = [&](size_t bytes) -> char* {
    char* r = p;
    p += (bytes + 255) & ~(size_t)255;
    return r;
  };
  __bf16* qb   = (__bf16*)take((size_t)B_ * H_ * N_ * DH_ * 2);
  __bf16* kb   = (__bf16*)take((size_t)B_ * H_ * N_ * DH_ * 2);
  __bf16* vT   = (__bf16*)take((size_t)B_ * H_ * DH_ * N_ * 2);
  __bf16* att  = (__bf16*)take((size_t)B_ * N_ * D_ * 2);
  __bf16* hb   = (__bf16*)take((size_t)B_ * N_ * D_ * 2);
  float*  tmp1 = (float*)take((size_t)B_ * N_ * D_ * 4);
  float*  hf   = (float*)take((size_t)B_ * N_ * D_ * 4);
  float*  tmp2 = (float*)take((size_t)B_ * N_ * D_ * 4);
  __bf16* rb   = (__bf16*)take((size_t)B_ * N_ * FF_ * 2);
  __bf16* xb   = (__bf16*)take((size_t)B_ * N_ * D_ * 2);
  __bf16* Wqb  = (__bf16*)take((size_t)D_ * D_ * 2);
  __bf16* Wkb  = (__bf16*)take((size_t)D_ * D_ * 2);
  __bf16* Wvb  = (__bf16*)take((size_t)D_ * D_ * 2);
  __bf16* Wob  = (__bf16*)take((size_t)D_ * D_ * 2);
  __bf16* W1b  = (__bf16*)take((size_t)FF_ * D_ * 2);
  __bf16* W2b  = (__bf16*)take((size_t)D_ * FF_ * 2);

  // one-time fp32 -> bf16 conversions (all L2-resident afterwards)
  const int nx8 = B_ * N_ * D_ / 8, nw8 = D_ * D_ / 8, nf8 = FF_ * D_ / 8;
  cvt_kernel<<<dim3((nx8 + 255) / 256), 256, 0, stream>>>(x, xb, nx8);
  cvt_kernel<<<dim3((nw8 + 255) / 256), 256, 0, stream>>>(Wq, Wqb, nw8);
  cvt_kernel<<<dim3((nw8 + 255) / 256), 256, 0, stream>>>(Wk, Wkb, nw8);
  cvt_kernel<<<dim3((nw8 + 255) / 256), 256, 0, stream>>>(Wv, Wvb, nw8);
  cvt_kernel<<<dim3((nw8 + 255) / 256), 256, 0, stream>>>(Wo, Wob, nw8);
  cvt_kernel<<<dim3((nf8 + 255) / 256), 256, 0, stream>>>(W1, W1b, nf8);
  cvt_kernel<<<dim3((nf8 + 255) / 256), 256, 0, stream>>>(W2, W2b, nf8);

  qkv_kernel<<<dim3(B_ * N_ / 16, 4, 3), 32, 0, stream>>>(
      xb, Wqb, bq, Wkb, bk, Wvb, bv, qb, kb, vT);
  qk_kernel<<<dim3(25 * 5, B_ * H_), 32, 0, stream>>>(qb, kb, sc_out);
  scores_kernel<<<dim3(N_, B_), 256, 0, stream>>>(sc_out, aw);
  av_kernel<<<dim3(25, B_ * H_), 32, 0, stream>>>(sc_out, vT, att);
  oproj_kernel<<<dim3(B_ * N_ / 16, 4), 32, 0, stream>>>(att, Wob, bo, x, tmp1);
  ln_kernel<<<dim3(B_ * N_), 256, 0, stream>>>(tmp1, g1, be1, hf, hb);
  ffn1_kernel<<<dim3(B_ * N_ / 16, 16), 32, 0, stream>>>(hb, W1b, b1, rb);
  ffn2_kernel<<<dim3(B_ * N_ / 16, 4), 32, 0, stream>>>(rb, W2b, b2, hf, tmp2);
  ln_kernel<<<dim3(B_ * N_), 256, 0, stream>>>(tmp2, g2, be2, y_out, (__bf16*)nullptr);

  (void)in_sizes; (void)n_in; (void)out_size; (void)ws_size;
}
